// EncoderText_37194416784057
// MI455X (gfx1250) — matrix operand order repaired and checked by reading, not compiled, not run on hardware
//
#include <hip/hip_runtime.h>

// ---------------- types ----------------
typedef __attribute__((ext_vector_type(16))) __bf16 v16bf;
typedef __attribute__((ext_vector_type(8)))  __bf16 v8bf;
typedef __attribute__((ext_vector_type(8)))  float  v8f;

// ---------------- problem constants ----------------
#define Bn 64
#define Ln 64
#define Dn 512
#define Hn 512
#define Sn 63           // merge steps
#define KDIM 1024       // 2*Dn
#define NSTR 528        // bf16 row stride for nsyn in LDS (pad, multiple of 8)

// output layout (floats)
#define FEATS_SZ  (Sn*Bn*Dn)            // 2064384
#define OWE_OFF   (3*FEATS_SZ)          // 6193152
#define TIDX_OFF  (OWE_OFF + Bn*Ln*Dn)  // 8290304
#define TPROB_OFF (TIDX_OFF + Sn*Bn)    // 8294336
#define SBND_OFF  (TPROB_OFF + Sn*Bn)   // 8298368

// ---------------- w1 -> bf16 WMMA-tile packing ----------------
// tile (kt,nt): 32(K) x 16(N). lane l holds 16 bf16: K = kt*32 + 16*(l/16) + j,
// N = nt*16 + (l%16). Packed: idx = ((kt*32+nt)*32 + lane)*16 + j  (32B/lane).
__global__ void pack_w1_bf16(const float* __restrict__ w1, void* __restrict__ w1bf_v) {
    __bf16* w1bf = (__bf16*)w1bf_v;
    int p = blockIdx.x * 256 + threadIdx.x;
    if (p >= KDIM * Hn) return;
    int tile   = p >> 9;
    int within = p & 511;
    int lane   = within >> 4;
    int j      = within & 15;
    int kt = tile >> 5, nt = tile & 31;
    int K = kt * 32 + ((lane >> 4) << 4) + j;
    int N = (nt << 4) + (lane & 15);
    w1bf[p] = (__bf16)w1[K * Hn + N];
}

// ---------------- main persistent kernel: 1 block per batch row ----------------
__global__ __launch_bounds__(256) void encoder_tree_kernel(
    const int*   __restrict__ x,
    const int*   __restrict__ lengths,
    const float* __restrict__ emb,
    const float* __restrict__ b1,
    const float* __restrict__ w2,
    const void*  __restrict__ w1bf_v,
    float*       __restrict__ syn_ws,   // [B][L][D] raw syn
    float*       __restrict__ sem_ws,   // [B][L][D] raw sem
    float*       __restrict__ out)
{
    const __bf16* __restrict__ w1bf = (const __bf16*)w1bf_v;

    __shared__ __align__(16) __bf16 nsyn[Ln * NSTR]; // normalized syn, bf16 (~67.6KB)
    __shared__ float partial[8][64];                 // per-wave logits partials
    __shared__ int   pos[80];
    __shared__ int   leftb[64], rightb[64];
    __shared__ float red4[8][4];
    __shared__ int   s_idx, s_slotA, s_slotB;
    __shared__ float s_undone;
    __shared__ float s_sc[6];

    const int b    = blockIdx.x;
    const int tid  = threadIdx.x;
    const int lane = tid & 31;
    const int wave = tid >> 5;
    const int len  = lengths[b];

    float* syn = syn_ws + (size_t)b * Ln * Dn;
    float* sem = sem_ws + (size_t)b * Ln * Dn;

    // lane-constant layer-2 weights (hoisted out of the 63-step loop)
    float w2v[4], b1v[4];
    #pragma unroll
    for (int ntl = 0; ntl < 4; ++ntl) {
        int n = (wave * 4 + ntl) * 16 + (lane & 15);
        w2v[ntl] = w2[n];
        b1v[ntl] = b1[n];
    }
    // lane-constant base of this wave's B-operand tiles
    const __bf16* __restrict__ bbase = w1bf + ((size_t)(wave * 4) << 9) + (lane << 4);

    // ---------- init: embedding gather, owe output, syn/sem, pos/bounds ----------
    for (int e = tid; e < Ln * Dn; e += 256) {
        int l = e >> 9, d = e & 511;
        float v = emb[(size_t)x[b * Ln + l] * Dn + d];
        syn[e] = v;
        sem[e] = v;
        out[OWE_OFF + (size_t)(b * Ln + l) * Dn + d] = (l < len) ? v : 0.0f;
    }
    if (tid < 80) pos[tid] = (tid < 63) ? tid : 63;
    if (tid < 64) { leftb[tid] = tid; rightb[tid] = tid; }
    __syncthreads();

    // per-row L2 norms -> nsyn (bf16)
    for (int l = wave; l < Ln; l += 8) {
        float s = 0.0f;
        for (int k = lane; k < Dn; k += 32) { float v = syn[l * Dn + k]; s += v * v; }
        for (int off = 16; off; off >>= 1) s += __shfl_xor(s, off, 32);
        float inv = 1.0f / (sqrtf(s) + 1e-8f);
        for (int k = lane; k < Dn; k += 32)
            nsyn[l * NSTR + k] = (__bf16)(syn[l * Dn + k] * inv);
    }
    __syncthreads();

    // ---------- 63 sequential merge steps ----------
    for (int it = 0; it < Sn; ++it) {
        const int M = Sn - it;                 // #candidate positions this step
        const int mtiles = (M + 15) >> 4;

        // ===== MLP via WMMA: partial[wave][m] = this wave's 64-col slice of
        //       relu(sf@w1 + b1) @ w2 =====
        for (int mt = 0; mt < mtiles; ++mt) {
            const int m    = mt * 16 + (lane & 15);
            const int rowL = pos[m + 1];       // k in [0,512): l2norm(syn[:,1:])
            const int rowR = pos[m];           // k in [512,1024): l2norm(syn[:,:-1])
            v8f acc[4] = {};
            for (int kt = 0; kt < 32; ++kt) {
                const int row = (kt < 16) ? rowL : rowR;
                const int kb  = ((kt & 15) << 5) + ((lane & 16) ? 16 : 0);
                const __bf16* ap = &nsyn[row * NSTR + kb];
                v8bf lo = *(const v8bf*)(ap);        // ds_load_b128
                v8bf hi = *(const v8bf*)(ap + 8);    // ds_load_b128
                v16bf a = __builtin_shufflevector(lo, hi,
                            0,1,2,3,4,5,6,7, 8,9,10,11,12,13,14,15);
                const __bf16* bp = bbase + ((size_t)kt << 14);   // kt*32*512 elems
                #pragma unroll
                for (int ntl = 0; ntl < 4; ++ntl) {
                    v16bf bf = *(const v16bf*)(bp + (ntl << 9));
                    acc[ntl] = __builtin_amdgcn_wmma_f32_16x16x32_bf16(
                        false, a, false, bf, (short)0, acc[ntl], false, false);
                }
            }
            // epilogue: fold the 4 N-tiles per lane first (sum over n splits
            // across tiles), giving 8 independent row-sums per lane; then run
            // the 16-lane reductions LEVEL-OUTER so the 8 ds_bpermute's per
            // level are independent and can be clause-grouped under one wait.
            float tot[8];
            #pragma unroll
            for (int r = 0; r < 8; ++r) {
                float t = 0.0f;
                #pragma unroll
                for (int ntl = 0; ntl < 4; ++ntl)
                    t += fmaxf(acc[ntl][r] + b1v[ntl], 0.0f) * w2v[ntl];
                tot[r] = t;
            }
            #pragma unroll
            for (int off = 8; off; off >>= 1) {
                float sh[8];
                #pragma unroll
                for (int r = 0; r < 8; ++r) sh[r] = __shfl_xor(tot[r], off, 32);
                #pragma unroll
                for (int r = 0; r < 8; ++r) tot[r] += sh[r];
            }
            if ((lane & 15) == 0) {
                const int mbase = mt * 16 + ((lane >> 4) << 3);
                #pragma unroll
                for (int r = 0; r < 8; ++r) partial[wave][mbase + r] = tot[r];
            }
        }
        __syncthreads();

        // ===== wave 0: sum partials, masked softmax + argmax, bookkeeping =====
        if (wave == 0) {
            int valid = len - 1 - it;
            valid = valid < 0 ? 0 : (valid > M ? M : valid);
            int p1i = lane, p2i = lane + 32;
            float l0 = 0.0f, l1 = 0.0f;
            #pragma unroll
            for (int w = 0; w < 8; ++w) { l0 += partial[w][p1i]; l1 += partial[w][p2i]; }
            float c0 = (p1i < M) ? l0 - ((p1i >= valid) ? 1e10f : 0.0f) : -3.0e38f;
            float c1 = (p2i < M) ? l1 - ((p2i >= valid) ? 1e10f : 0.0f) : -3.0e38f;
            float mv; int mi;
            if (c1 > c0) { mv = c1; mi = p2i; } else { mv = c0; mi = p1i; }
            for (int off = 16; off; off >>= 1) {
                float om = __shfl_xor(mv, off, 32);
                int   oi = __shfl_xor(mi, off, 32);
                if (om > mv || (om == mv && oi < mi)) { mv = om; mi = oi; }
            }
            float e0  = (p1i < M) ? __expf(c0 - mv) : 0.0f;
            float e1  = (p2i < M) ? __expf(c1 - mv) : 0.0f;
            float den = e0 + e1;
            for (int off = 16; off; off >>= 1) den += __shfl_xor(den, off, 32);
            if (lane == 0) {
                int idx = mi;
                int sA = pos[idx], sB = pos[idx + 1];
                s_idx = idx; s_slotA = sA; s_slotB = sB;
                s_undone = (it < len - 1) ? 1.0f : 0.0f;
                out[TIDX_OFF  + it * Bn + b] = (float)idx;
                out[TPROB_OFF + it * Bn + b] = 1.0f / den;
                out[SBND_OFF + (size_t)(it * Bn + b) * 2 + 0] = (float)leftb[sA];
                out[SBND_OFF + (size_t)(it * Bn + b) * 2 + 1] = (float)rightb[sB];
                rightb[sA] = rightb[sB];
            }
        }
        __syncthreads();

        // ===== merge: norms of fl, fr, fl+fr, synA+synB =====
        const int sA = s_slotA, sB = s_slotB;
        float q0 = 0, q1 = 0, q2 = 0, q3 = 0;
        for (int d = tid; d < Dn; d += 256) {
            float fl = sem[sA * Dn + d], fr = sem[sB * Dn + d];
            float s  = fl + fr;
            float ys = syn[sA * Dn + d] + syn[sB * Dn + d];
            q0 += fl * fl; q1 += fr * fr; q2 += s * s; q3 += ys * ys;
        }
        for (int off = 16; off; off >>= 1) {
            q0 += __shfl_xor(q0, off, 32);
            q1 += __shfl_xor(q1, off, 32);
            q2 += __shfl_xor(q2, off, 32);
            q3 += __shfl_xor(q3, off, 32);
        }
        if (lane == 0) { red4[wave][0] = q0; red4[wave][1] = q1; red4[wave][2] = q2; red4[wave][3] = q3; }
        __syncthreads();
        if (tid == 0) {
            float n0 = 0, n1 = 0, n2 = 0, n3 = 0;
            for (int w = 0; w < 8; ++w) { n0 += red4[w][0]; n1 += red4[w][1]; n2 += red4[w][2]; n3 += red4[w][3]; }
            const float eps = 1e-8f;
            float nfl = sqrtf(n0), nfr = sqrtf(n1), ns = sqrtf(n2), nys = sqrtf(n3);
            float undone = s_undone;
            float inv_s = 1.0f / (ns + eps);
            float nf2   = ns * inv_s;
            s_sc[0] = inv_s;                                    // merged sem row
            s_sc[1] = inv_s * (1.0f / (nf2 + eps)) * undone;    // feats scale on raw (fl+fr)
            s_sc[2] = (1.0f / (nfl + eps)) * undone;            // lfeats
            s_sc[3] = (1.0f / (nfr + eps)) * undone;            // rfeats
            float inv_y = 1.0f / (nys + eps);
            float nfy   = nys * inv_y;
            s_sc[4] = inv_y;                                    // merged syn row
            s_sc[5] = inv_y * (1.0f / (nfy + eps));             // merged nsyn row (normalized)
        }
        __syncthreads();

        // ===== write outputs + merged rows; shift pos =====
        {
            const float scSem = s_sc[0], scF = s_sc[1], scL = s_sc[2], scR = s_sc[3];
            const float scSyn = s_sc[4], scN = s_sc[5];
            for (int d = tid; d < Dn; d += 256) {
                float fl = sem[sA * Dn + d], fr = sem[sB * Dn + d];
                float s  = fl + fr;
                float ys = syn[sA * Dn + d] + syn[sB * Dn + d];
                size_t o = (size_t)(it * Bn + b) * Dn + d;
                out[o]                = s  * scF;
                out[FEATS_SZ + o]     = fl * scL;
                out[2 * FEATS_SZ + o] = fr * scR;
                sem[sA * Dn + d] = s  * scSem;
                syn[sA * Dn + d] = ys * scSyn;
                nsyn[sA * NSTR + d] = (__bf16)(ys * scN);
            }
            if (tid == 0) {
                for (int p = s_idx + 1; p < 63; ++p) pos[p] = pos[p + 1];
            }
        }
        __syncthreads();
    }
}

// ---------------- host launch ----------------
extern "C" void kernel_launch(void* const* d_in, const int* in_sizes, int n_in,
                              void* d_out, int out_size, void* d_ws, size_t ws_size,
                              hipStream_t stream) {
    const int*   x       = (const int*)d_in[0];
    const int*   lengths = (const int*)d_in[1];
    const float* emb     = (const float*)d_in[2];
    const float* w1      = (const float*)d_in[3];
    const float* b1      = (const float*)d_in[4];
    const float* w2      = (const float*)d_in[5];
    float*       out     = (float*)d_out;

    // workspace: [0,1MB) packed bf16 w1; then raw syn (8MB); then raw sem (8MB)
    void*  w1bf   = d_ws;
    float* syn_ws = (float*)((char*)d_ws + (size_t)1024 * 1024);
    float* sem_ws = syn_ws + (size_t)Bn * Ln * Dn;

    pack_w1_bf16<<<(KDIM * Hn + 255) / 256, 256, 0, stream>>>(w1, w1bf);
    encoder_tree_kernel<<<Bn, 256, 0, stream>>>(x, lengths, emb, b1, w2, w1bf,
                                                syn_ws, sem_ws, out);
}
